// GCN_36258113913469
// MI455X (gfx1250) — matrix operand order repaired and checked by reading, not compiled, not run on hardware
//
#include <hip/hip_runtime.h>
#include <hip/hip_bf16.h>

// ---------------------------------------------------------------------------
// GCN (3-layer) for MI455X / gfx1250, wave32 + WMMA bf16.
//   GEMMs:  v_wmma_f32_16x16x32_bf16; 8 waves/block, NTILES 16x16 tiles/wave.
//           Packed-bf16 B slice staged once per block in LDS (shared by all
//           waves) -> inner loop reads B via ds_load_b128, A packed with
//           v_cvt_pk_bf16_f32.
//   SpMM :  edge-parallel float4 gather + hardware f32 atomics into L2.
// ---------------------------------------------------------------------------

typedef __attribute__((ext_vector_type(16))) __bf16 v16bf;
typedef __attribute__((ext_vector_type(8)))  float  v8f;
typedef __attribute__((ext_vector_type(2)))  __bf16 v2bf;

union Frag {
  v16bf    v;
  unsigned u[8];
  uint4    q[2];
};

__device__ __forceinline__ unsigned pack_bf16(float a, float b) {
#if __has_builtin(__builtin_amdgcn_cvt_pk_bf16_f32)
  v2bf r = __builtin_amdgcn_cvt_pk_bf16_f32(a, b);   // 1 VALU, RNE
  return __builtin_bit_cast(unsigned, r);
#else
  // hardware packed f32->bf16 convert (RNE); 1 VALU per dword
  unsigned r;
  asm("v_cvt_pk_bf16_f32 %0, %1, %2" : "=v"(r) : "v"(a), "v"(b));
  return r;
#endif
}

// ---------------------------------------------------------------------------
// Utility kernels
// ---------------------------------------------------------------------------
__global__ void zero_kernel(float* __restrict__ p, size_t n) {
  size_t i = (size_t)blockIdx.x * blockDim.x + threadIdx.x;
  size_t stride = (size_t)gridDim.x * blockDim.x;
  for (; i < n; i += stride) p[i] = 0.0f;
}

__global__ void deg_kernel(const int* __restrict__ src, const int* __restrict__ dst,
                           float* __restrict__ odeg, float* __restrict__ ideg, int E) {
  int e = blockIdx.x * blockDim.x + threadIdx.x;
  if (e < E) {
    unsafeAtomicAdd(&odeg[src[e]], 1.0f);
    unsafeAtomicAdd(&ideg[dst[e]], 1.0f);
  }
}

__global__ void norm_kernel(float* __restrict__ p, size_t n) {
  size_t i = (size_t)blockIdx.x * blockDim.x + threadIdx.x;
  if (i < n) p[i] = rsqrtf(fmaxf(p[i], 1.0f));   // clip(deg,1)^-0.5
}

// ---------------------------------------------------------------------------
// Pre-pack W[K x NCOL] (f32 row-major) into bf16 WMMA B-fragment layout:
//   Wp[((tileN*NCHUNK + chunk)*32 + lane)*8 + v]
// lane&15 = column-in-tile, lane>>4 selects K half (0/16),
// dword v holds K = chunk*32 + kb + 2v (lo) and +1 (hi).
// Out-of-range columns (NCOL=40 tail) are zero-filled -> no guards in GEMM.
// ---------------------------------------------------------------------------
template <int K, int NCOL, int NT>
__global__ void pack_w_kernel(const float* __restrict__ W, unsigned* __restrict__ Wp) {
  constexpr int NCHUNK = K / 32;
  int idx = blockIdx.x * blockDim.x + threadIdx.x;   // (tile, chunk, lane)
  if (idx >= NT * NCHUNK * 32) return;
  int lane  = idx & 31;
  int chunk = (idx >> 5) % NCHUNK;
  int tile  = (idx >> 5) / NCHUNK;
  int n  = tile * 16 + (lane & 15);
  int kb = (lane >> 4) * 16;
  bool ok = (n < NCOL);
  unsigned* out = Wp + (size_t)idx * 8;
#pragma unroll
  for (int v = 0; v < 8; ++v) {
    int k = chunk * 32 + kb + 2 * v;
    float b0 = ok ? W[(size_t)k * NCOL + n] : 0.0f;
    float b1 = ok ? W[(size_t)(k + 1) * NCOL + n] : 0.0f;
    out[v] = pack_bf16(b0, b1);
  }
}

// ---------------------------------------------------------------------------
// WMMA GEMM:  out[M x NCOL] = epilogue( A[M x K] @ W[K x NCOL] )
// B slice for this block's column group staged in LDS (shared by all waves).
//   SCALEA : scale A rows by onorm[row] (layer 1)
//   EPI==0 : plain store
//   EPI==1 : relu(y*inorm[row] + bias[col]) * onorm[row]
// ---------------------------------------------------------------------------
template <int K, int NCOL, int LDA, int NTILES, bool SCALEA, int EPI>
__global__ __launch_bounds__(256)
void gemm_wmma(const float* __restrict__ A, const unsigned* __restrict__ Wp,
               const float* __restrict__ bias, const float* __restrict__ onorm,
               const float* __restrict__ inorm, float* __restrict__ out, int nrows) {
  constexpr int NCHUNK = K / 32;
  constexpr int SLICEQ = NTILES * NCHUNK * 64;     // B slice size in uint4
  __shared__ uint4 smem[SLICEQ];

  const int wave = threadIdx.x >> 5;
  const int lane = threadIdx.x & 31;
  const int tileM = blockIdx.x * 8 + wave;
  const int tileN0 = blockIdx.y * NTILES;

  // --- cooperative fill: packed-B slice for this column group -> LDS ---
  {
    const uint4* gsrc = (const uint4*)(Wp + (size_t)tileN0 * NCHUNK * 256);
#pragma unroll
    for (int i = threadIdx.x; i < SLICEQ; i += 256) smem[i] = gsrc[i];
  }
  __syncthreads();

  if (tileM * 16 < nrows) {                        // wave-uniform guard
    const int m   = lane & 15;
    const int hi  = lane >> 4;
    const int row = tileM * 16 + m;                // A row for this lane
    const int kbA = hi * 16;                       // A half-wave K split (0/16)

    const float scaleA = SCALEA ? onorm[row] : 1.0f;
    const float* ap = A + (size_t)row * LDA;

    v8f c[NTILES];
#pragma unroll
    for (int t = 0; t < NTILES; ++t) c[t] = {};

#pragma unroll
    for (int k0 = 0; k0 < K; k0 += 32) {
      const int chunk = k0 >> 5;
      // --- A fragment: 4x b128 loads + v_cvt_pk_bf16_f32 pack ---
      Frag a;
      const float4* pa0 = (const float4*)(ap + k0 + kbA);
      const float4* pa1 = (const float4*)(ap + k0 + kbA + 8);
      float4 f0 = pa0[0], f1 = pa0[1];
      float4 f2 = pa1[0], f3 = pa1[1];
      if (SCALEA) {
        f0.x *= scaleA; f0.y *= scaleA; f0.z *= scaleA; f0.w *= scaleA;
        f1.x *= scaleA; f1.y *= scaleA; f1.z *= scaleA; f1.w *= scaleA;
        f2.x *= scaleA; f2.y *= scaleA; f2.z *= scaleA; f2.w *= scaleA;
        f3.x *= scaleA; f3.y *= scaleA; f3.z *= scaleA; f3.w *= scaleA;
      }
      a.u[0] = pack_bf16(f0.x, f0.y); a.u[1] = pack_bf16(f0.z, f0.w);
      a.u[2] = pack_bf16(f1.x, f1.y); a.u[3] = pack_bf16(f1.z, f1.w);
      a.u[4] = pack_bf16(f2.x, f2.y); a.u[5] = pack_bf16(f2.z, f2.w);
      a.u[6] = pack_bf16(f3.x, f3.y); a.u[7] = pack_bf16(f3.z, f3.w);

      if (k0 + 32 < K) __builtin_prefetch(ap + k0 + 32, 0, 3);

#pragma unroll
      for (int t = 0; t < NTILES; ++t) {           // NTILES WMMAs, one A frag
        Frag b;                                    // B fragment from LDS
        const uint4* bt = &smem[((t * NCHUNK + chunk) * 32 + lane) * 2];
        b.q[0] = bt[0];
        b.q[1] = bt[1];
        c[t] = __builtin_amdgcn_wmma_f32_16x16x32_bf16(
            /*neg_a=*/false, a.v, /*neg_b=*/false, b.v,
            /*c_mod=*/(short)0, c[t], /*reuse_a=*/false, /*reuse_b=*/false);
      }
    }

    // D layout: lane half selects M block (0..7 / 8..15), N = lane&15
    const int rbase = tileM * 16 + hi * 8;
#pragma unroll
    for (int r = 0; r < 8; ++r) {
      const int orow = rbase + r;
      float inr = 0.0f, onr = 0.0f;
      if (EPI == 1) { inr = inorm[orow]; onr = onorm[orow]; }
#pragma unroll
      for (int t = 0; t < NTILES; ++t) {
        int   col = (tileN0 + t) * 16 + m;
        float y   = c[t][r];
        if (EPI == 1) y = fmaxf(y * inr + bias[col], 0.0f) * onr;
        if (col < NCOL) out[(size_t)orow * NCOL + col] = y;
      }
    }
  }
}

// ---------------------------------------------------------------------------
// SpMM: agg[dst] += h[src] * ew   (segment_sum of u_mul_e)
// ---------------------------------------------------------------------------
template <int F, int GL2>
__global__ __launch_bounds__(256)
void spmm_kernel(const float* __restrict__ h, const int* __restrict__ src,
                 const int* __restrict__ dst, const float* __restrict__ ew,
                 float* __restrict__ agg, int E) {
  const int G = F / 4;
  long long idx = (long long)blockIdx.x * blockDim.x + threadIdx.x;
  int       g   = (int)(idx & ((1 << GL2) - 1));
  long long e   = idx >> GL2;
  if (e >= E || g >= G) return;
  int   s = src[e], d = dst[e];
  float w = ew[e];
  float4 v = *(const float4*)(h + (size_t)s * F + g * 4);
  float* p = agg + (size_t)d * F + g * 4;
  unsafeAtomicAdd(p + 0, v.x * w);
  unsafeAtomicAdd(p + 1, v.y * w);
  unsafeAtomicAdd(p + 2, v.z * w);
  unsafeAtomicAdd(p + 3, v.w * w);
}

// post layer-1: x = relu(agg*in_norm + b1) * out_norm   (out_norm folded fwd)
__global__ void post1_kernel(const float* __restrict__ agg, const float* __restrict__ b,
                             const float* __restrict__ inorm, const float* __restrict__ onorm,
                             float* __restrict__ out, size_t total) {
  size_t i = (size_t)blockIdx.x * blockDim.x + threadIdx.x;
  if (i >= total) return;
  int row = (int)(i >> 7), col = (int)(i & 127);
  out[i] = fmaxf(agg[i] * inorm[row] + b[col], 0.0f) * onorm[row];
}

// final: out = agg*in_norm + b3  (no activation)
__global__ void post3_kernel(const float* __restrict__ agg, const float* __restrict__ b,
                             const float* __restrict__ inorm, float* __restrict__ out,
                             size_t total) {
  size_t i = (size_t)blockIdx.x * blockDim.x + threadIdx.x;
  if (i >= total) return;
  int row = (int)(i / 40), col = (int)(i - (size_t)row * 40);
  out[i] = agg[i] * inorm[row] + b[col];
}

// ---------------------------------------------------------------------------
// Launcher
// ---------------------------------------------------------------------------
extern "C" void kernel_launch(void* const* d_in, const int* in_sizes, int n_in,
                              void* d_out, int out_size, void* d_ws, size_t ws_size,
                              hipStream_t stream) {
  const float* features = (const float*)d_in[0];
  const int*   src      = (const int*)d_in[1];
  const int*   dst      = (const int*)d_in[2];
  const float* ew       = (const float*)d_in[3];
  const float* W1       = (const float*)d_in[4];
  const float* b1       = (const float*)d_in[5];
  const float* W2       = (const float*)d_in[6];
  const float* b2       = (const float*)d_in[7];
  const float* W3       = (const float*)d_in[8];
  const float* b3       = (const float*)d_in[9];

  const int E = in_sizes[1];
  const int N = in_sizes[0] / 256;                 // F_IN = 256

  float*       ws = (float*)d_ws;
  const size_t NP = 51200;                         // N padded for alignment
  float* onorm = ws;                               // deg -> norm, in place
  float* inorm = ws + NP;
  float* B0 = ws + 2 * NP;                         // N x 128 (h1 / X1 / X2)
  float* B1 = B0 + (size_t)N * 128;                // N x 128 (agg1 / agg2)
  float* B2 = B1 + (size_t)N * 128;                // N x 40  (T3)
  float* B3 = B2 + (size_t)N * 40;                 // N x 40  (agg3)
  unsigned* Wp1 = (unsigned*)(B3 + (size_t)N * 40);        // 8*8*256  dwords
  unsigned* Wp2 = Wp1 + 8 * 8 * 256;                       // 8*4*256  dwords
  unsigned* Wp3 = Wp2 + 8 * 4 * 256;                       // 3*4*256  dwords

  const int tilesM = (N + 15) / 16;                // 3125
  const dim3 gemmBlk(256);
  const dim3 gemmGrd128((tilesM + 7) / 8, 2);      // 128 cols = 2 x (4 tiles)
  const dim3 gemmGrd40((tilesM + 7) / 8, 1);       // 40 cols  = 1 x (3 tiles)

  const size_t nf128 = (size_t)N * 128;
  const size_t nf40  = (size_t)N * 40;
  auto blocks = [](size_t n) { return (unsigned)((n + 255) / 256); };

  // degrees -> norms; pack weights to bf16 fragments
  zero_kernel<<<blocks(2 * NP), 256, 0, stream>>>(onorm, 2 * NP);
  deg_kernel<<<blocks((size_t)E), 256, 0, stream>>>(src, dst, onorm, inorm, E);
  norm_kernel<<<blocks(2 * NP), 256, 0, stream>>>(onorm, 2 * NP);
  pack_w_kernel<256, 128, 8><<<blocks(8 * 8 * 32), 256, 0, stream>>>(W1, Wp1);
  pack_w_kernel<128, 128, 8><<<blocks(8 * 4 * 32), 256, 0, stream>>>(W2, Wp2);
  pack_w_kernel<128, 40, 3><<<blocks(3 * 4 * 32), 256, 0, stream>>>(W3, Wp3);

  // ----- layer 1: (features * out_norm) @ W1 -> SpMM -> relu(*in_norm+b1)*out_norm
  gemm_wmma<256, 128, 256, 4, true, 0>
      <<<gemmGrd128, gemmBlk, 0, stream>>>(features, Wp1, b1, onorm, inorm, B0, N);
  zero_kernel<<<blocks(nf128), 256, 0, stream>>>(B1, nf128);
  spmm_kernel<128, 5><<<blocks((size_t)E * 32), 256, 0, stream>>>(B0, src, dst, ew, B1, E);
  post1_kernel<<<blocks(nf128), 256, 0, stream>>>(B1, b1, inorm, onorm, B0, nf128);

  // ----- layer 2: SpMM -> @W2 with fused relu(*in_norm+b2)*out_norm epilogue
  zero_kernel<<<blocks(nf128), 256, 0, stream>>>(B1, nf128);
  spmm_kernel<128, 5><<<blocks((size_t)E * 32), 256, 0, stream>>>(B0, src, dst, ew, B1, E);
  gemm_wmma<128, 128, 128, 4, false, 1>
      <<<gemmGrd128, gemmBlk, 0, stream>>>(B1, Wp2, b2, onorm, inorm, B0, N);

  // ----- layer 3: @W3 -> SpMM -> *in_norm + b3
  gemm_wmma<128, 40, 128, 3, false, 0>
      <<<gemmGrd40, gemmBlk, 0, stream>>>(B0, Wp3, b3, onorm, inorm, B2, N);
  zero_kernel<<<blocks(nf40), 256, 0, stream>>>(B3, nf40);
  spmm_kernel<40, 4><<<blocks((size_t)E * 16), 256, 0, stream>>>(B2, src, dst, ew, B3, E);
  post3_kernel<<<blocks(nf40), 256, 0, stream>>>(B3, b3, inorm, (float*)d_out, nf40);
}
